// BahdanauAttention_86328842649813
// MI455X (gfx1250) — compile-verified, matrix operand order checked by reading, not run on hardware
//
#include <hip/hip_runtime.h>
#include <hip/hip_bf16.h>

// ---------------------------------------------------------------------------
// Bahdanau attention, fused for MI455X (gfx1250, wave32, WMMA bf16).
// B=32, T=1024, H=1024.
// ---------------------------------------------------------------------------

#define BB 32
#define TT 1024
#define HH 1024

typedef __attribute__((ext_vector_type(16))) __bf16 v16bf;
typedef __attribute__((ext_vector_type(8)))  float  v8f;

__device__ __forceinline__ unsigned short f32_to_bf16(float f) {
  union { float f; unsigned int u; } v; v.f = f;
  unsigned int u = v.u;
  u += 0x7FFFu + ((u >> 16) & 1u);   // round-to-nearest-even
  return (unsigned short)(u >> 16);
}

__device__ __forceinline__ uint2 pack4_bf16(float4 s) {
  uint2 p;
  p.x = (unsigned)f32_to_bf16(s.x) | ((unsigned)f32_to_bf16(s.y) << 16);
  p.y = (unsigned)f32_to_bf16(s.z) | ((unsigned)f32_to_bf16(s.w) << 16);
  return p;
}

// Fragment-layout helpers for 16-bit 16x32 WMMA operand (ISA 7.12.2):
//   lane = (row & 15) + 16 * ((k >> 3) & 1)
//   elem i = ((k & 16) >> 1) | (k & 7)          (k = column within 32-block)
// For k aligned to 4, elements i..i+3 are contiguous bf16 -> one 8B store.

// ---------------------------------------------------------------------------
// Kernel 1: h_proj[b,k] = sum_h h_t[b,h] * Wa[k,h]     (32 x 1024, tiny)
// ---------------------------------------------------------------------------
__global__ __launch_bounds__(256) void hproj_kernel(const float* __restrict__ h_t,
                                                    const float* __restrict__ Wa,
                                                    float* __restrict__ h_proj) {
  const int k = blockIdx.x * 256 + threadIdx.x;     // 0..1023
  __shared__ float sh[BB][64];
  float acc[BB];
#pragma unroll
  for (int b = 0; b < BB; ++b) acc[b] = 0.f;

  for (int hc = 0; hc < HH; hc += 64) {
    __syncthreads();
    for (int i = threadIdx.x; i < BB * 64; i += 256) {
      int b = i >> 6, h = i & 63;
      sh[b][h] = h_t[b * HH + hc + h];
    }
    __syncthreads();
    const float* wrow = Wa + (size_t)k * HH + hc;
    for (int h = 0; h < 64; ++h) {
      float w = wrow[h];
#pragma unroll
      for (int b = 0; b < BB; ++b) acc[b] += sh[b][h] * w;
    }
  }
#pragma unroll
  for (int b = 0; b < BB; ++b) h_proj[b * HH + k] = acc[b];
}

// ---------------------------------------------------------------------------
// Kernel 2: convert Ua (f32, [1024][1024], row = output k, col = h) into bf16
// WMMA-B fragment order: [n_tile(64)][kb(32)][lane(32)][elem(16)]   (2 MB)
// ---------------------------------------------------------------------------
__global__ __launch_bounds__(256) void ua_convert_kernel(const float* __restrict__ Ua,
                                                         unsigned short* __restrict__ uaf) {
  const int idx  = blockIdx.x * 256 + threadIdx.x;  // 0..262143
  const int kout = idx >> 8;                        // Ua row (0..1023)
  const int h    = (idx & 255) << 2;                // column, 4-aligned
  float4 s = *(const float4*)(Ua + (size_t)kout * HH + h);
  uint2 p = pack4_bf16(s);
  const int nt   = kout >> 4;
  const int kb   = h >> 5;
  const int lane = (kout & 15) + (((h >> 3) & 1) << 4);
  const int i0   = ((h & 16) >> 1) | (h & 4);
  size_t off = (((size_t)(nt * 32 + kb)) * 32 + lane) * 16 + i0;
  *(uint2*)(uaf + off) = p;
}

// ---------------------------------------------------------------------------
// Kernel 3 (hot): fused
//   e[b,t] = sum_k va[k] * tanh(h_proj[b,k] + enc_out[b,t,:]·Ua[k,:])
// One WG (256 thr = 8 waves) per 64 flattened rows = 4 M-tiles of 16.
// A-tile (64x1024) in bf16 fragment order in LDS (128 KB), reused by all 64
// N-tiles.  Each wave: 8 N-tiles x 32 kb-steps; per kb-step one B-fragment
// load feeds 4 v_wmma_f32_16x16x32_bf16 (4x B reuse -> 4x less Ua L2 traffic).
// ---------------------------------------------------------------------------
__global__ __launch_bounds__(256) void e_kernel(const float* __restrict__ enc_out,
                                                const float* __restrict__ h_proj,
                                                const float* __restrict__ va,
                                                const unsigned short* __restrict__ uaf,
                                                const unsigned char* __restrict__ src_mask,
                                                float* __restrict__ e_out) {
  __shared__ __align__(32) unsigned short aFrag[4][32][32][16]; // [mt][kb][lane][elem] 128KB
  __shared__ float e_tile[64];

  const int tile = blockIdx.x;          // 0..511
  const int row0 = tile * 64;           // flattened (b*T + t); 64 | T so b fixed
  const int b    = row0 >> 10;          // T = 1024
  const int tid  = threadIdx.x;

  if (tid < 64) e_tile[tid] = 0.f;

  // --- load + convert A tile (64 rows x 1024) into fragment-ordered LDS ---
  for (int it = 0; it < 64; ++it) {
    int idx = tid + it * 256;           // 0..16383
    int m = idx >> 8;                   // row in tile (0..63)
    int k = (idx & 255) << 2;           // 4-aligned column
    float4 s = *(const float4*)(enc_out + (size_t)(row0 + m) * HH + k);
    uint2 p = pack4_bf16(s);
    int mt   = m >> 4;
    int kb   = k >> 5;
    int lane = (m & 15) + (((k >> 3) & 1) << 4);
    int i0   = ((k & 16) >> 1) | (k & 4);
    *(uint2*)&aFrag[mt][kb][lane][i0] = p;
  }
  __syncthreads();

  const int wave = tid >> 5;
  const int lane = tid & 31;
  const v16bf* __restrict__ bfrag = (const v16bf*)uaf;

  float vacc[4][8];
#pragma unroll
  for (int mt = 0; mt < 4; ++mt)
#pragma unroll
    for (int r = 0; r < 8; ++r) vacc[mt][r] = 0.f;

  for (int nt = wave; nt < 64; nt += 8) {        // 8 N-tiles per wave
    const int ncol = nt * 16 + (lane & 15);      // global output column k
    const float hp = h_proj[b * HH + ncol];
    const float vv = va[ncol];

    v8f c[4];
#pragma unroll
    for (int mt = 0; mt < 4; ++mt)
      c[mt] = (v8f){0.f, 0.f, 0.f, 0.f, 0.f, 0.f, 0.f, 0.f};

    const v16bf* bp = bfrag + (size_t)nt * 1024; // 32 kb-blocks * 32 lanes
#pragma unroll 2
    for (int kb = 0; kb < 32; ++kb) {
      v16bf bv = bp[kb * 32 + lane];
      __builtin_prefetch((const void*)(bp + (kb + 8) * 32 + lane), 0, 0);
#pragma unroll
      for (int mt = 0; mt < 4; ++mt) {
        v16bf a = *(const v16bf*)&aFrag[mt][kb][lane][0];
        c[mt] = __builtin_amdgcn_wmma_f32_16x16x32_bf16(
                    /*neg_a=*/false, a, /*neg_b=*/false, bv,
                    /*c_mod=*/(short)0, c[mt], /*reuse_a=*/false, /*reuse_b=*/false);
      }
    }
    // C layout: lane holds column n = lane&15 ; VGPR r -> row m = r + 8*(lane>>4)
#pragma unroll
    for (int mt = 0; mt < 4; ++mt)
#pragma unroll
      for (int r = 0; r < 8; ++r)
        vacc[mt][r] += tanhf(c[mt][r] + hp) * vv;
  }

  // reduce across the 16 lanes sharing each row group (half-wave groups)
#pragma unroll
  for (int mt = 0; mt < 4; ++mt)
#pragma unroll
    for (int r = 0; r < 8; ++r) {
      float s = vacc[mt][r];
      for (int off = 8; off > 0; off >>= 1) s += __shfl_xor(s, off, 16);
      if ((lane & 15) == 0)
        atomicAdd(&e_tile[mt * 16 + r + 8 * (lane >> 4)], s);
    }
  __syncthreads();

  if (tid < 64) {
    int gi = row0 + tid;                 // = b*T + t
    float ev = e_tile[tid];
    e_out[gi] = src_mask[gi] ? ev : -__builtin_inff();
  }
}

// ---------------------------------------------------------------------------
// Kernel 4: row softmax over T per batch.  One WG per b.
// ---------------------------------------------------------------------------
__global__ __launch_bounds__(256) void softmax_kernel(const float* __restrict__ e,
                                                      float* __restrict__ attn) {
  const int b = blockIdx.x, tid = threadIdx.x;
  __shared__ float sred[8];
  const float* er = e + b * TT;
  float vals[4];
  float m = -__builtin_inff();
#pragma unroll
  for (int i = 0; i < 4; ++i) { vals[i] = er[tid + i * 256]; m = fmaxf(m, vals[i]); }
  for (int off = 16; off > 0; off >>= 1) m = fmaxf(m, __shfl_xor(m, off, 32));
  if ((tid & 31) == 0) sred[tid >> 5] = m;
  __syncthreads();
  float bm = sred[0];
#pragma unroll
  for (int j = 1; j < 8; ++j) bm = fmaxf(bm, sred[j]);
  __syncthreads();

  float s = 0.f;
#pragma unroll
  for (int i = 0; i < 4; ++i) { float ex = __expf(vals[i] - bm); vals[i] = ex; s += ex; }
  for (int off = 16; off > 0; off >>= 1) s += __shfl_xor(s, off, 32);
  if ((tid & 31) == 0) sred[tid >> 5] = s;
  __syncthreads();
  float bs = 0.f;
#pragma unroll
  for (int j = 0; j < 8; ++j) bs += sred[j];
  float inv = 1.0f / bs;
#pragma unroll
  for (int i = 0; i < 4; ++i) attn[b * TT + tid + i * 256] = vals[i] * inv;
}

// ---------------------------------------------------------------------------
// Kernel 5: context[b,h] = sum_t attn[b,t] * enc_out[b,t,h]   (bandwidth bound)
// ---------------------------------------------------------------------------
__global__ __launch_bounds__(256) void context_kernel(const float* __restrict__ enc_out,
                                                      const float* __restrict__ attn,
                                                      float* __restrict__ ctx) {
  const int b = blockIdx.y;
  const int h = blockIdx.x * 256 + threadIdx.x;
  const float* __restrict__ ep = enc_out + (size_t)b * TT * HH + h;
  const float* __restrict__ ap = attn + b * TT;
  float acc = 0.f;
  for (int t = 0; t < TT; t += 4) {
    float a0 = ap[t], a1 = ap[t + 1], a2 = ap[t + 2], a3 = ap[t + 3];
    acc += a0 * ep[(size_t)(t    ) * HH];
    acc += a1 * ep[(size_t)(t + 1) * HH];
    acc += a2 * ep[(size_t)(t + 2) * HH];
    acc += a3 * ep[(size_t)(t + 3) * HH];
  }
  ctx[b * HH + h] = acc;
}

// ---------------------------------------------------------------------------
extern "C" void kernel_launch(void* const* d_in, const int* in_sizes, int n_in,
                              void* d_out, int out_size, void* d_ws, size_t ws_size,
                              hipStream_t stream) {
  const float*         h_t      = (const float*)d_in[0];
  const float*         enc_out  = (const float*)d_in[1];
  const unsigned char* src_mask = (const unsigned char*)d_in[2];
  const float*         Wa       = (const float*)d_in[3];
  const float*         Ua       = (const float*)d_in[4];
  const float*         va       = (const float*)d_in[5];

  float* ctx_out  = (float*)d_out;                 // [B,H]
  float* attn_out = (float*)d_out + BB * HH;       // [B,T]

  char* ws = (char*)d_ws;
  float*          h_proj = (float*)ws;                              // 128 KB
  float*          e_buf  = (float*)(ws + BB * HH * 4);              // 128 KB
  unsigned short* uaf    = (unsigned short*)(ws + 2 * BB * HH * 4); // 2 MB bf16 frags

  hproj_kernel     <<<dim3(HH / 256),          256, 0, stream>>>(h_t, Wa, h_proj);
  ua_convert_kernel<<<dim3(HH * HH / 4 / 256), 256, 0, stream>>>(Ua, uaf);
  e_kernel         <<<dim3(BB * TT / 64),      256, 0, stream>>>(enc_out, h_proj, va, uaf,
                                                                 src_mask, e_buf);
  softmax_kernel   <<<dim3(BB),                256, 0, stream>>>(e_buf, attn_out);
  context_kernel   <<<dim3(HH / 256, BB),      256, 0, stream>>>(enc_out, attn_out, ctx_out);
}
